// TemporalAttention_84971632984366
// MI455X (gfx1250) — matrix operand order repaired and checked by reading, not compiled, not run on hardware
//
#include <hip/hip_runtime.h>
#include <hip/hip_bf16.h>

typedef __attribute__((ext_vector_type(16))) __bf16       v16bf;
typedef __attribute__((ext_vector_type(8)))  float        v8f;
typedef __attribute__((ext_vector_type(8)))  unsigned int v8u;

constexpr int B_ = 64;
constexpr int T_ = 8192;
constexpr int D_ = 128;
constexpr int U_ = 128;
constexpr int TILE_TOK = 64;      // tokens per block in kernel 1
constexpr int XS = D_ + 8;        // padded LDS row stride (bf16 elems)

// ---------------------------------------------------------------------------
// Kernel 1: logits[b*T+t] = V . tanh(x_bt W + Wb) + Vb
// 256 threads = 8 waves; wave(w): token sub-tile tw=w>>1 (16 tokens),
// unit half nh=w&1 (64 units). Each wave: 4 K-steps x 4 N-tiles = 16 WMMAs.
// ---------------------------------------------------------------------------
__global__ void __launch_bounds__(256)
ta_logits_kernel(const float* __restrict__ x, const float* __restrict__ Wk,
                 const float* __restrict__ Wb, const float* __restrict__ Vk,
                 const float* __restrict__ Vb, float* __restrict__ logits)
{
    __shared__ __bf16 sX[TILE_TOK][XS];   // x tile, bf16, [token][d]
    __shared__ __bf16 sWt[U_][XS];        // W transposed, bf16: sWt[u][d] = W[d][u]
    __shared__ float  sV[U_];
    __shared__ float  sWb[U_];
    __shared__ float  sLogit[TILE_TOK];

    const int tid  = threadIdx.x;
    const int tok0 = blockIdx.x * TILE_TOK;

    // ---- stage x tile (fp32 -> bf16), coalesced ----
#pragma unroll
    for (int i = 0; i < (TILE_TOK * D_) / 256; ++i) {
        int idx = tid + 256 * i;
        int r = idx >> 7, c = idx & 127;
        sX[r][c] = (__bf16)x[(size_t)(tok0 + r) * D_ + c];
    }
    // ---- stage W transposed (fp32 -> bf16); W is [D][U] row-major ----
#pragma unroll
    for (int i = 0; i < (D_ * U_) / 256; ++i) {
        int idx = tid + 256 * i;
        int d = idx >> 7, u = idx & 127;
        sWt[u][d] = (__bf16)Wk[idx];
    }
    if (tid < U_)       { sV[tid] = Vk[tid]; sWb[tid] = Wb[tid]; }
    if (tid < TILE_TOK) sLogit[tid] = 0.0f;
    __syncthreads();

    const int wave = tid >> 5;
    const int lane = tid & 31;
    const int tw   = wave >> 1;     // token sub-tile 0..3
    const int nh   = wave & 1;      // unit half 0..1
    const int n16  = lane & 15;
    const int lh   = lane >> 4;     // lane half

    // A-fragment row for this lane (A: M=16 rows striped over lanes 0..15)
    const int m = tw * 16 + n16;
    const unsigned int* arow = (const unsigned int*)(&sX[m][0]);

    const v8f vzero = {0.f,0.f,0.f,0.f,0.f,0.f,0.f,0.f};
    v8f acc[4];
#pragma unroll
    for (int nt = 0; nt < 4; ++nt) acc[nt] = vzero;

#pragma unroll
    for (int kb = 0; kb < D_; kb += 32) {
        // A frag 16x32 bf16: vgpr v holds K pair; pairIdx = kb/2 + (v<4 ? v : 4+v) + 4*lh
        v8u ap;
#pragma unroll
        for (int v = 0; v < 8; ++v) {
            int pidx = (kb >> 1) + ((v < 4) ? v : (4 + v)) + 4 * lh;
            ap[v] = arow[pidx];
        }
        v16bf afrag = __builtin_bit_cast(v16bf, ap);

#pragma unroll
        for (int nt = 0; nt < 4; ++nt) {
            // B frag 32x16 bf16: lane = N col; pairIdx = kb/2 + (v<4 ? v : 4+v) + 4*lh (same as A)
            int col = nh * 64 + nt * 16 + n16;
            const unsigned int* brow = (const unsigned int*)(&sWt[col][0]);
            v8u bp;
#pragma unroll
            for (int v = 0; v < 8; ++v)
                bp[v] = brow[(kb >> 1) + ((v < 4) ? v : (4 + v)) + 4 * lh];
            v16bf bfrag = __builtin_bit_cast(v16bf, bp);

            acc[nt] = __builtin_amdgcn_wmma_f32_16x16x32_bf16(
                false, afrag, false, bfrag, (short)0, acc[nt], false, false);
        }
    }

    // tanh + dot with V: C layout -> lane holds N=n16, rows r+8*lh
    float p[8] = {0.f,0.f,0.f,0.f,0.f,0.f,0.f,0.f};
#pragma unroll
    for (int nt = 0; nt < 4; ++nt) {
        int col = nh * 64 + nt * 16 + n16;
        float vs = sV[col], wb = sWb[col];
#pragma unroll
        for (int r = 0; r < 8; ++r)
            p[r] += tanhf(acc[nt][r] + wb) * vs;
    }
    // reduce over the 16 N-lanes within each lane-half
#pragma unroll
    for (int r = 0; r < 8; ++r) {
        float v = p[r];
        v += __shfl_xor(v, 1, 16);
        v += __shfl_xor(v, 2, 16);
        v += __shfl_xor(v, 4, 16);
        v += __shfl_xor(v, 8, 16);
        p[r] = v;
    }
    if (n16 == 0) {
        int mb = tw * 16 + 8 * lh;
#pragma unroll
        for (int r = 0; r < 8; ++r)
            atomicAdd(&sLogit[mb + r], p[r]);   // combine the two unit-halves
    }
    __syncthreads();

    if (tid < TILE_TOK)
        logits[tok0 + tid] = sLogit[tid] + Vb[0];
}

// ---------------------------------------------------------------------------
// Kernel 2: per-batch softmax over T + context = sum_t w_t * x_bt
// One block per b; 512 threads = 4 t-groups x 128 d-lanes.
// ---------------------------------------------------------------------------
__global__ void __launch_bounds__(512)
ta_softmax_ctx_kernel(const float* __restrict__ x, float* __restrict__ logits,
                      float* __restrict__ out)
{
    __shared__ float sred[512];
    const int b   = blockIdx.x;
    const int tid = threadIdx.x;
    float* lg = logits + (size_t)b * T_;

    // phase 1: max over T
    float lm = -3.402823466e38f;
    for (int i = tid; i < T_; i += 512) lm = fmaxf(lm, lg[i]);
    sred[tid] = lm; __syncthreads();
    for (int s = 256; s > 0; s >>= 1) {
        if (tid < s) sred[tid] = fmaxf(sred[tid], sred[tid + s]);
        __syncthreads();
    }
    float mx = sred[0]; __syncthreads();

    // phase 2: exp in place + sum
    float ls = 0.0f;
    for (int i = tid; i < T_; i += 512) {
        float e = __expf(lg[i] - mx);
        lg[i] = e;
        ls += e;
    }
    sred[tid] = ls; __syncthreads();
    for (int s = 256; s > 0; s >>= 1) {
        if (tid < s) sred[tid] += sred[tid + s];
        __syncthreads();
    }
    float inv = 1.0f / sred[0]; __syncthreads();

    // phase 3: weighted sum over T (second streaming pass over x)
    const int d  = tid & 127;
    const int tg = tid >> 7;
    const float* xb = x + (size_t)b * T_ * D_;
    float acc = 0.0f;
    for (int t = tg; t < T_; t += 4)
        acc += lg[t] * xb[(size_t)t * D_ + d];
    sred[tid] = acc; __syncthreads();
    if (tid < 128) {
        float tot = sred[d] + sred[128 + d] + sred[256 + d] + sred[384 + d];
        out[b * D_ + d] = tot * inv;
    }
}

// ---------------------------------------------------------------------------
extern "C" void kernel_launch(void* const* d_in, const int* in_sizes, int n_in,
                              void* d_out, int out_size, void* d_ws, size_t ws_size,
                              hipStream_t stream)
{
    const float* x  = (const float*)d_in[0];   // [B,T,D]
    const float* Wk = (const float*)d_in[1];   // [D,U]
    const float* Wb = (const float*)d_in[2];   // [U]
    const float* Vk = (const float*)d_in[3];   // [U,1]
    const float* Vb = (const float*)d_in[4];   // [1]
    float* out    = (float*)d_out;             // [B,D]
    float* logits = (float*)d_ws;              // B*T floats = 2 MB scratch

    ta_logits_kernel<<<(B_ * T_) / TILE_TOK, 256, 0, stream>>>(
        x, Wk, Wb, Vk, Vb, logits);
    ta_softmax_ctx_kernel<<<B_, 512, 0, stream>>>(x, logits, out);
}
